// DiffusionNetwork_80187039416641
// MI455X (gfx1250) — matrix operand order repaired and checked by reading, not compiled, not run on hardware
//
#include <hip/hip_runtime.h>
#include <cmath>

// Problem dims (fixed by the reference)
#define BB 16384
#define DD 2048
#define AA 64
#define NSTEPS 30

typedef __attribute__((ext_vector_type(8)))  float  v8f;
typedef __attribute__((ext_vector_type(16))) __bf16 v16bf;

struct StepConsts {
    float c1[NSTEPS];  // (1-alpha[t]) / sqrt(1-alpha_bar[t])
    float c2[NSTEPS];  // 1 / sqrt(alpha[t])
    float c3[NSTEPS];  // t>0 ? sqrt(beta[t]) : 0
};

__device__ __forceinline__ float gelu_exact(float x) {
    return 0.5f * x * (1.0f + erff(x * 0.70710678118654752440f));
}

union AFrag  { v16bf v; uint4 q[2]; };
union BFrag  { v16bf v; uint4 q[2]; };
union U4BF   { uint4 q; __bf16 h[8]; };

// ---------------------------------------------------------------------------
// Kernel 0a/0b: transpose f32 matrix [R,C] -> bf16 matrix [C,R]
// ---------------------------------------------------------------------------
__global__ __launch_bounds__(256)
void transpose_to_bf16(const float* __restrict__ in, __bf16* __restrict__ out,
                       int R, int C) {
    __shared__ float tile[32][33];
    int bx = blockIdx.x * 32;   // col offset in `in`
    int by = blockIdx.y * 32;   // row offset in `in`
    int tx = threadIdx.x, ty = threadIdx.y;
    for (int i = ty; i < 32; i += 8) {
        int r = by + i, c = bx + tx;
        tile[i][tx] = (r < R && c < C) ? in[(size_t)r * C + c] : 0.0f;
    }
    __syncthreads();
    for (int i = ty; i < 32; i += 8) {
        int c = bx + i, r = by + tx;  // out[c][r] = in[r][c]
        if (c < C && r < R) out[(size_t)c * R + r] = (__bf16)tile[tx][i];
    }
}

// ---------------------------------------------------------------------------
// Kernel 0c: ct[t][n] = b1[n] + sum_k time_embed[t][k] * W1[k][n]
// ---------------------------------------------------------------------------
__global__ __launch_bounds__(256)
void ct_kernel(const float* __restrict__ te, const float* __restrict__ W1,
               const float* __restrict__ b1, float* __restrict__ ct) {
    int idx = blockIdx.x * blockDim.x + threadIdx.x;
    if (idx >= NSTEPS * DD) return;
    int t = idx / DD, n = idx % DD;
    float s = b1[n];
    const float* ter = te + (size_t)t * DD;
    for (int k = 0; k < DD; ++k)
        s = fmaf(ter[k], W1[(size_t)k * DD + n], s);   // te broadcast, W1 coalesced in n
    ct[idx] = s;
}

// ---------------------------------------------------------------------------
// Kernel 1: Z1b(bf16)[B,D] = (bf16)( z(f32)[B,D] @ W1 )  via W1tb = W1^T (bf16)
// One wave: 16 rows x 64 cols (4 WMMA accumulators), K-loop over 2048.
// Block = 8 waves -> 128 rows x 64 cols.
// ---------------------------------------------------------------------------
__global__ __launch_bounds__(256)
void gemm1_kernel(const float* __restrict__ z, const __bf16* __restrict__ W1tb,
                  __bf16* __restrict__ Z1b) {
    int lane = threadIdx.x & 31;
    int wv   = threadIdx.x >> 5;
    int m0   = blockIdx.x * 128 + wv * 16;
    int n0   = blockIdx.y * 64;
    int hi   = lane >> 4;          // half-wave select
    int lm   = lane & 15;
    int m    = m0 + lm;            // A-matrix row held by this lane

    v8f acc0 = {}, acc1 = {}, acc2 = {}, acc3 = {};
    const float* zr = z + (size_t)m * DD;

    for (int k0 = 0; k0 < DD; k0 += 32) {
        // ---- A fragment (16x32 bf16): elems 0-7 = K kb..kb+7, 8-15 = K kb+16..kb+23
        int kb = k0 + hi * 8;
        float fa[8], fb[8];
        *(float4*)&fa[0] = *(const float4*)(zr + kb);
        *(float4*)&fa[4] = *(const float4*)(zr + kb + 4);
        *(float4*)&fb[0] = *(const float4*)(zr + kb + 16);
        *(float4*)&fb[4] = *(const float4*)(zr + kb + 20);
        AFrag a;
#pragma unroll
        for (int i = 0; i < 8; ++i) {
            a.v[i]     = (__bf16)fa[i];   // native v_cvt_pk_bf16_f32 packing
            a.v[8 + i] = (__bf16)fb[i];
        }
        // ---- B fragments (32x16 bf16): lane holds col n, K kb2..kb2+7 and kb2+16..kb2+23
        int kb2 = k0 + hi * 8;
#pragma unroll
        for (int nt = 0; nt < 4; ++nt) {
            const uint4* bp = (const uint4*)(W1tb + (size_t)(n0 + nt * 16 + lm) * DD + kb2);
            BFrag b; b.q[0] = bp[0]; b.q[1] = bp[2];
            v8f* accp = (nt == 0) ? &acc0 : (nt == 1) ? &acc1 : (nt == 2) ? &acc2 : &acc3;
            *accp = __builtin_amdgcn_wmma_f32_16x16x32_bf16(
                        false, a.v, false, b.v, (short)0, *accp, false, false);
        }
    }
    // ---- store C: VGPR j -> row m0 + j + 8*hi, col n0 + nt*16 + lm
#pragma unroll
    for (int nt = 0; nt < 4; ++nt) {
        v8f acc = (nt == 0) ? acc0 : (nt == 1) ? acc1 : (nt == 2) ? acc2 : acc3;
#pragma unroll
        for (int j = 0; j < 8; ++j) {
            int row = m0 + j + hi * 8;
            int col = n0 + nt * 16 + lm;
            Z1b[(size_t)row * DD + col] = (__bf16)acc[j];
        }
    }
}

// ---------------------------------------------------------------------------
// Kernel 2: all 30 diffusion steps fused. Each wave owns rows m0..m0+15 and
// all 64 action columns (action kept in registers). Per step: fused
// gelu(Z1 + ct[t]) -> bf16 A-frags -> WMMA vs W2^T -> action update.
// Z1b (64 MB bf16) stays L2-resident across the 30 re-reads.
// ---------------------------------------------------------------------------
__global__ __launch_bounds__(128)
void diffusion_steps_kernel(const __bf16* __restrict__ Z1b,
                            const __bf16* __restrict__ W2tb,
                            const float* __restrict__ ct,
                            const float* __restrict__ b2,
                            const float* __restrict__ init_noise,
                            const float* __restrict__ step_noise,
                            float* __restrict__ out, StepConsts sc) {
    int lane = threadIdx.x & 31;
    int wv   = threadIdx.x >> 5;
    int m0   = blockIdx.x * 64 + wv * 16;
    int hi   = lane >> 4;
    int lm   = lane & 15;
    int m    = m0 + lm;

    // action registers, laid out to match the WMMA C fragment
    float act[4][8];
#pragma unroll
    for (int nt = 0; nt < 4; ++nt)
#pragma unroll
        for (int j = 0; j < 8; ++j) {
            int row = m0 + j + hi * 8;
            int col = nt * 16 + lm;
            act[nt][j] = init_noise[(size_t)row * AA + col];
        }
    float b2v[4];
#pragma unroll
    for (int nt = 0; nt < 4; ++nt) b2v[nt] = b2[nt * 16 + lm];

    const __bf16* z1r = Z1b + (size_t)m * DD;

    for (int t = NSTEPS - 1; t >= 0; --t) {
        const float* ctt = ct + (size_t)t * DD;
        v8f acc[4] = {};
        for (int k0 = 0; k0 < DD; k0 += 32) {
            // A fragment: gelu(Z1 + ct[t]) fused on the fly
            int kb = k0 + hi * 8;
            U4BF z0, z1;
            z0.q = *(const uint4*)(z1r + kb);          // K kb..kb+7
            z1.q = *(const uint4*)(z1r + kb + 16);     // K kb+16..kb+23
            float c0[8], c1[8];
            *(float4*)&c0[0] = *(const float4*)(ctt + kb);
            *(float4*)&c0[4] = *(const float4*)(ctt + kb + 4);
            *(float4*)&c1[0] = *(const float4*)(ctt + kb + 16);
            *(float4*)&c1[4] = *(const float4*)(ctt + kb + 20);
            AFrag a;
#pragma unroll
            for (int i = 0; i < 8; ++i) {
                a.v[i]     = (__bf16)gelu_exact((float)z0.h[i] + c0[i]);
                a.v[8 + i] = (__bf16)gelu_exact((float)z1.h[i] + c1[i]);
            }
            int kb2 = k0 + hi * 8;
#pragma unroll
            for (int nt = 0; nt < 4; ++nt) {
                const uint4* bp = (const uint4*)(W2tb + (size_t)(nt * 16 + lm) * DD + kb2);
                BFrag b; b.q[0] = bp[0]; b.q[1] = bp[2];
                acc[nt] = __builtin_amdgcn_wmma_f32_16x16x32_bf16(
                              false, a.v, false, b.v, (short)0, acc[nt], false, false);
            }
        }
        // action update: a = (a - c1*pred) * c2 + c3 * noise[t]
        float k1 = sc.c1[t], k2 = sc.c2[t], k3 = sc.c3[t];
#pragma unroll
        for (int nt = 0; nt < 4; ++nt)
#pragma unroll
            for (int j = 0; j < 8; ++j) {
                int row = m0 + j + hi * 8;
                int col = nt * 16 + lm;
                float pred = acc[nt][j] + b2v[nt];
                float nse  = step_noise[((size_t)t * BB + row) * AA + col];
                act[nt][j] = (act[nt][j] - k1 * pred) * k2 + k3 * nse;
            }
    }
#pragma unroll
    for (int nt = 0; nt < 4; ++nt)
#pragma unroll
        for (int j = 0; j < 8; ++j) {
            int row = m0 + j + hi * 8;
            int col = nt * 16 + lm;
            out[(size_t)row * AA + col] = act[nt][j];
        }
}

// ---------------------------------------------------------------------------
extern "C" void kernel_launch(void* const* d_in, const int* in_sizes, int n_in,
                              void* d_out, int out_size, void* d_ws, size_t ws_size,
                              hipStream_t stream) {
    const float* z          = (const float*)d_in[0];   // [B,D]
    const float* time_embed = (const float*)d_in[1];   // [STEPS,D]
    const float* W1         = (const float*)d_in[2];   // [D,D]
    const float* b1         = (const float*)d_in[3];   // [D]
    const float* W2         = (const float*)d_in[4];   // [D,A]
    const float* b2         = (const float*)d_in[5];   // [A]
    const float* init_noise = (const float*)d_in[6];   // [B,A]
    const float* step_noise = (const float*)d_in[7];   // [STEPS,B,A]
    float* out = (float*)d_out;

    // Workspace carve-up (all sizes 256B-aligned already)
    char* ws = (char*)d_ws;
    __bf16* Z1b  = (__bf16*)ws;       ws += (size_t)BB * DD * 2;     // 64 MB
    __bf16* W1tb = (__bf16*)ws;       ws += (size_t)DD * DD * 2;     // 8 MB
    __bf16* W2tb = (__bf16*)ws;       ws += (size_t)AA * DD * 2;     // 256 KB
    float*  ctb  = (float*)ws;        ws += (size_t)NSTEPS * DD * 4; // 240 KB

    // Host-side cosine schedule (doubles, deterministic)
    double ab[NSTEPS + 1];
    for (int i = 0; i <= NSTEPS; ++i) {
        double tt = (double)i / NSTEPS;
        double v  = std::cos((tt + 0.008) / 1.008 * M_PI / 2.0);
        ab[i] = v * v;
    }
    double ab0 = ab[0];
    for (int i = 0; i <= NSTEPS; ++i) ab[i] /= ab0;
    StepConsts sc;
    double abar = 1.0;
    for (int t = 0; t < NSTEPS; ++t) {
        double beta = 1.0 - ab[t + 1] / ab[t];
        if (beta < 0.0) beta = 0.0; if (beta > 0.999) beta = 0.999;
        double alpha = 1.0 - beta;
        abar *= alpha;
        sc.c1[t] = (float)((1.0 - alpha) / std::sqrt(1.0 - abar));
        sc.c2[t] = (float)(1.0 / std::sqrt(alpha));
        sc.c3[t] = (t > 0) ? (float)std::sqrt(beta) : 0.0f;
    }

    // 0a/0b: transposed bf16 weights
    transpose_to_bf16<<<dim3(DD / 32, DD / 32), dim3(32, 8), 0, stream>>>(W1, W1tb, DD, DD);
    transpose_to_bf16<<<dim3(AA / 32, DD / 32), dim3(32, 8), 0, stream>>>(W2, W2tb, DD, AA);
    // 0c: per-step conditioning row ct[t] = te[t]@W1 + b1
    ct_kernel<<<(NSTEPS * DD + 255) / 256, 256, 0, stream>>>(time_embed, W1, b1, ctb);
    // 1: Z1 = z @ W1 (WMMA bf16, f32 accum), stored bf16 (L2-resident)
    gemm1_kernel<<<dim3(BB / 128, DD / 64), 256, 0, stream>>>(z, W1tb, Z1b);
    // 2: all 30 steps fused (gelu + skinny WMMA GEMM + action recurrence)
    diffusion_steps_kernel<<<dim3(BB / 64), 128, 0, stream>>>(
        Z1b, W2tb, ctb, b2, init_noise, step_noise, out, sc);
}
